// BitMultiheadAttention158Inference_38354057953286
// MI455X (gfx1250) — 16-bit and 8-bit matrix operand order repaired and checked by reading, compiled, not run on hardware
//
#include <hip/hip_runtime.h>
#include <hip/hip_bf16.h>

typedef __attribute__((ext_vector_type(2)))  int      v2i;
typedef __attribute__((ext_vector_type(4)))  int      v4i;
typedef __attribute__((ext_vector_type(8)))  int      v8i;
typedef __attribute__((ext_vector_type(8)))  float    v8f;
typedef __attribute__((ext_vector_type(16))) _Float16 v16h;

#define NBATCH 4
#define SEQ    2048
#define EMB    1024
#define NHEAD  16
#define HDIM   64
#define NTOK   (NBATCH * SEQ)
#define WELEM  (EMB * EMB)

union FragI8  { v8i  v; v2i d[4]; v4i q[2]; };
union FragF16 { v16h v; v4i q[2]; };


__device__ inline v8i load_a_i8(const signed char* base, int row, int k0, int lane) {
    const signed char* p = base + (size_t)row * EMB + k0 + ((lane >> 4) * 16);
    FragI8 f;
    f.d[0] = *(const v2i*)(p);
    f.d[1] = *(const v2i*)(p + 8);
    f.d[2] = *(const v2i*)(p + 32);
    f.d[3] = *(const v2i*)(p + 40);
    return f.v;
}
__device__ inline v8i load_b_i8(const signed char* base, int col, int k0, int lane) {
    const signed char* p = base + (size_t)col * EMB + k0 + ((lane >> 4) * 16);
    FragI8 f;
    f.q[0] = *(const v4i*)(p);
    f.q[1] = *(const v4i*)(p + 32);
    return f.v;
}
__device__ inline v16h load_a_f16(const _Float16* rowp, int lane) {
    int kb = (lane >> 4) * 8;
    FragF16 f;
    f.q[0] = *(const v4i*)(rowp + kb);
    f.q[1] = *(const v4i*)(rowp + kb + 16);
    return f.v;
}
__device__ inline v16h load_b_f16(const _Float16* colp, int lane) {
    int kb = (lane >> 4) * 8;
    FragF16 f;
    f.q[0] = *(const v4i*)(colp + kb);
    f.q[1] = *(const v4i*)(colp + kb + 16);
    return f.v;
}


__global__ __launch_bounds__(256) void k_wabs_partial(const float* __restrict__ w_in,
                                                      const float* __restrict__ w_out,
                                                      float* __restrict__ partials) {
    __shared__ float red[256];
    int t   = blockIdx.x >> 8;
    int blk = blockIdx.x & 255;
    const float* src = (t < 3) ? (w_in + (size_t)t * WELEM) : w_out;
    int base = blk * 4096;
    float s = 0.f;
    for (int i = threadIdx.x; i < 4096; i += 256) s += fabsf(src[base + i]);
    red[threadIdx.x] = s; __syncthreads();
    for (int off = 128; off > 0; off >>= 1) {
        if (threadIdx.x < off) red[threadIdx.x] += red[threadIdx.x + off];
        __syncthreads();
    }
    if (threadIdx.x == 0) partials[t * 256 + blk] = red[0];
}

__global__ __launch_bounds__(256) void k_wscale_final(const float* __restrict__ partials,
                                                      float* __restrict__ wscale) {
    __shared__ float red[256];
    int t = blockIdx.x;
    red[threadIdx.x] = partials[t * 256 + threadIdx.x]; __syncthreads();
    for (int off = 128; off > 0; off >>= 1) {
        if (threadIdx.x < off) red[threadIdx.x] += red[threadIdx.x + off];
        __syncthreads();
    }
    if (threadIdx.x == 0) wscale[t] = fmaxf(red[0] * (1.0f / (float)WELEM), 1e-5f);
}


__global__ __launch_bounds__(256) void k_quant_w(const float* __restrict__ w_in,
                                                 const float* __restrict__ w_out,
                                                 const float* __restrict__ wscale,
                                                 signed char* __restrict__ wq) {
    int idx = blockIdx.x * 256 + threadIdx.x;
    int t = idx >> 20;
    float w = (t < 3) ? w_in[idx] : w_out[idx - 3 * WELEM];
    float r = rintf(w / wscale[t]);
    r = fminf(fmaxf(r, -1.f), 1.f);
    wq[idx] = (signed char)r;
}


__global__ __launch_bounds__(256) void k_quant_act(const float* __restrict__ src,
                                                   signed char* __restrict__ dst,
                                                   float* __restrict__ gamma_out) {
    __shared__ float red[256];
    int row = blockIdx.x;
    const float* sp = src + (size_t)row * EMB;
    float m = 0.f;
    for (int i = threadIdx.x; i < EMB; i += 256) m = fmaxf(m, fabsf(sp[i]));
    red[threadIdx.x] = m; __syncthreads();
    for (int off = 128; off > 0; off >>= 1) {
        if (threadIdx.x < off) red[threadIdx.x] = fmaxf(red[threadIdx.x], red[threadIdx.x + off]);
        __syncthreads();
    }
    float gamma = fmaxf(red[0], 1e-5f);
    if (threadIdx.x == 0) gamma_out[row] = gamma;
    float qs = 128.0f / gamma;
    signed char* dp = dst + (size_t)row * EMB;
    for (int i = threadIdx.x; i < EMB; i += 256) {
        float r = rintf(sp[i] * qs);
        r = fminf(fmaxf(r, -128.f), 127.f);
        dp[i] = (signed char)r;
    }
}


__global__ __launch_bounds__(128) void k_gemm_qkv(const signed char* __restrict__ actq,
                                                  const float* __restrict__ gamma,
                                                  const signed char* __restrict__ wq,
                                                  const float* __restrict__ wscale,
                                                  const float* __restrict__ bias,
                                                  _Float16* __restrict__ Qh,
                                                  _Float16* __restrict__ Kh,
                                                  _Float16* __restrict__ Vt) {
    int lane = threadIdx.x & 31;
    int wave = threadIdx.x >> 5;
    int gw = blockIdx.x * 4 + wave;
    int nt = gw % 48, mt = gw / 48;
    int m0 = mt * 32, n0 = nt * 64;
    int t = n0 >> 10;
    const signed char* A = actq + (size_t)t * NTOK * EMB;
    const float* g = gamma + t * NTOK;
    int laneLo = lane & 15, laneHi = lane >> 4;

    v8i zi = {0, 0, 0, 0, 0, 0, 0, 0};
    v8i acc[2][4];
#pragma unroll
    for (int mi = 0; mi < 2; ++mi)
#pragma unroll
        for (int ni = 0; ni < 4; ++ni) acc[mi][ni] = zi;

    for (int k0 = 0; k0 < EMB; k0 += 64) {
        v8i af[2], bf[4];
#pragma unroll
        for (int mi = 0; mi < 2; ++mi)
            af[mi] = load_a_i8(A, m0 + mi * 16 + laneLo, k0, lane);
#pragma unroll
        for (int ni = 0; ni < 4; ++ni)
            bf[ni] = load_b_i8(wq, n0 + ni * 16 + laneLo, k0, lane);
#pragma unroll
        for (int mi = 0; mi < 2; ++mi)
#pragma unroll
            for (int ni = 0; ni < 4; ++ni)
                acc[mi][ni] = __builtin_amdgcn_wmma_i32_16x16x64_iu8(
                    true, af[mi], true, bf[ni], acc[mi][ni], false, false);
    }

    float wst = wscale[t] * (1.0f / 128.0f);
#pragma unroll
    for (int mi = 0; mi < 2; ++mi)
#pragma unroll
        for (int ni = 0; ni < 4; ++ni) {
            int n  = n0 + ni * 16 + laneLo;
            int r  = n & 1023, hd = r >> 6, d = r & 63;
            float bv = bias[n];
#pragma unroll
            for (int v = 0; v < 8; ++v) {
                int m = m0 + mi * 16 + v + 8 * laneHi;
                float out = (float)acc[mi][ni][v] * (wst * g[m]) + bv;
                int b = m >> 11, l = m & 2047;
                if (t == 0) {
                    Qh[((size_t)(b * NHEAD + hd) * SEQ + l) * HDIM + d] =
                        (_Float16)(0.125f * out);
                } else if (t == 1) {
                    Kh[((size_t)(b * NHEAD + hd) * SEQ + l) * HDIM + d] = (_Float16)out;
                } else {
                    Vt[((size_t)(b * NHEAD + hd) * HDIM + d) * SEQ + l] = (_Float16)out;
                }
            }
        }
}


__global__ __launch_bounds__(128) void k_attention(const _Float16* __restrict__ Qh,
                                                   const _Float16* __restrict__ Kh,
                                                   const _Float16* __restrict__ Vt,
                                                   float* __restrict__ ctx) {
    __shared__ __align__(16) _Float16 Pbuf[4][32][32];
    int lane = threadIdx.x & 31;
    int wave = threadIdx.x >> 5;
    int gw = blockIdx.x * 4 + wave;
    int bh = gw >> 6;
    int q0 = (gw & 63) * 32;
    const _Float16* Qb = Qh + (size_t)bh * SEQ * HDIM;
    const _Float16* Kb = Kh + (size_t)bh * SEQ * HDIM;
    const _Float16* Vb = Vt + (size_t)bh * HDIM * SEQ;
    int laneLo = lane & 15, laneHi = lane >> 4;

    v16h qf[2][2];
#pragma unroll
    for (int mi = 0; mi < 2; ++mi)
#pragma unroll
        for (int c = 0; c < 2; ++c)
            qf[mi][c] = load_a_f16(Qb + (size_t)(q0 + mi * 16 + laneLo) * HDIM + c * 32, lane);

    v8f zf = {0.f, 0.f, 0.f, 0.f, 0.f, 0.f, 0.f, 0.f};
    v8f acc[2][4];
    float mrow[2][8], lrow[2][8];
#pragma unroll
    for (int mi = 0; mi < 2; ++mi) {
#pragma unroll
        for (int ni = 0; ni < 4; ++ni) acc[mi][ni] = zf;
#pragma unroll
        for (int v = 0; v < 8; ++v) { mrow[mi][v] = -1e30f; lrow[mi][v] = 0.f; }
    }

    for (int k0 = 0; k0 < SEQ; k0 += 32) {
        v16h kf[2][2];
#pragma unroll
        for (int si = 0; si < 2; ++si)
#pragma unroll
            for (int c = 0; c < 2; ++c)
                kf[si][c] = load_b_f16(Kb + (size_t)(k0 + si * 16 + laneLo) * HDIM + c * 32, lane);

#pragma unroll
        for (int mi = 0; mi < 2; ++mi) {
            v8f s0 = zf, s1 = zf;
            s0 = __builtin_amdgcn_wmma_f32_16x16x32_f16(false, qf[mi][0], false, kf[0][0], (short)0, s0, false, false);
            s0 = __builtin_amdgcn_wmma_f32_16x16x32_f16(false, qf[mi][1], false, kf[0][1], (short)0, s0, false, false);
            s1 = __builtin_amdgcn_wmma_f32_16x16x32_f16(false, qf[mi][0], false, kf[1][0], (short)0, s1, false, false);
            s1 = __builtin_amdgcn_wmma_f32_16x16x32_f16(false, qf[mi][1], false, kf[1][1], (short)0, s1, false, false);

#pragma unroll
            for (int v = 0; v < 8; ++v) {
                float mx = fmaxf(s0[v], s1[v]);
#pragma unroll
                for (int off = 8; off > 0; off >>= 1)
                    mx = fmaxf(mx, __shfl_xor(mx, off, 32));
                float mo = mrow[mi][v];
                float mn = fmaxf(mo, mx);
                float alpha = __expf(mo - mn);
                mrow[mi][v] = mn;
                float p0 = __expf(s0[v] - mn);
                float p1 = __expf(s1[v] - mn);
                float rs = p0 + p1;
#pragma unroll
                for (int off = 8; off > 0; off >>= 1)
                    rs += __shfl_xor(rs, off, 32);
                lrow[mi][v] = lrow[mi][v] * alpha + rs;
#pragma unroll
                for (int ni = 0; ni < 4; ++ni) acc[mi][ni][v] *= alpha;
                int row = mi * 16 + v + 8 * laneHi;
                Pbuf[wave][row][laneLo]      = (_Float16)p0;
                Pbuf[wave][row][laneLo + 16] = (_Float16)p1;
            }
        }

        v16h pf[2];
#pragma unroll
        for (int mi = 0; mi < 2; ++mi) {
            const _Float16* pr = &Pbuf[wave][mi * 16 + laneLo][0];
            int kb = laneHi * 8;
            FragF16 f;
            f.q[0] = *(const v4i*)(pr + kb);
            f.q[1] = *(const v4i*)(pr + kb + 16);
            pf[mi] = f.v;
        }
        v16h vf[4];
#pragma unroll
        for (int ni = 0; ni < 4; ++ni)
            vf[ni] = load_b_f16(Vb + (size_t)(ni * 16 + laneLo) * SEQ + k0, lane);

#pragma unroll
        for (int mi = 0; mi < 2; ++mi)
#pragma unroll
            for (int ni = 0; ni < 4; ++ni)
                acc[mi][ni] = __builtin_amdgcn_wmma_f32_16x16x32_f16(
                    false, pf[mi], false, vf[ni], (short)0, acc[mi][ni], false, false);
    }

    int b = bh >> 4, h = bh & 15;
#pragma unroll
    for (int mi = 0; mi < 2; ++mi)
#pragma unroll
        for (int ni = 0; ni < 4; ++ni)
#pragma unroll
            for (int v = 0; v < 8; ++v) {
                int row = q0 + mi * 16 + v + 8 * laneHi;
                int tok = b * SEQ + row;
                int col = h * HDIM + ni * 16 + laneLo;
                ctx[(size_t)tok * EMB + col] = acc[mi][ni][v] / lrow[mi][v];
            }
}


__global__ __launch_bounds__(128) void k_gemm_out(const signed char* __restrict__ actq,
                                                  const float* __restrict__ gamma,
                                                  const signed char* __restrict__ wq,
                                                  const float* __restrict__ wscale,
                                                  const float* __restrict__ bias,
                                                  float* __restrict__ out) {
    int lane = threadIdx.x & 31;
    int wave = threadIdx.x >> 5;
    int gw = blockIdx.x * 4 + wave;
    int nt = gw % 16, mt = gw / 16;
    int m0 = mt * 32, n0 = nt * 64;
    int laneLo = lane & 15, laneHi = lane >> 4;

    v8i zi = {0, 0, 0, 0, 0, 0, 0, 0};
    v8i acc[2][4];
#pragma unroll
    for (int mi = 0; mi < 2; ++mi)
#pragma unroll
        for (int ni = 0; ni < 4; ++ni) acc[mi][ni] = zi;

    for (int k0 = 0; k0 < EMB; k0 += 64) {
        v8i af[2], bf[4];
#pragma unroll
        for (int mi = 0; mi < 2; ++mi)
            af[mi] = load_a_i8(actq, m0 + mi * 16 + laneLo, k0, lane);
#pragma unroll
        for (int ni = 0; ni < 4; ++ni)
            bf[ni] = load_b_i8(wq, n0 + ni * 16 + laneLo, k0, lane);
#pragma unroll
        for (int mi = 0; mi < 2; ++mi)
#pragma unroll
            for (int ni = 0; ni < 4; ++ni)
                acc[mi][ni] = __builtin_amdgcn_wmma_i32_16x16x64_iu8(
                    true, af[mi], true, bf[ni], acc[mi][ni], false, false);
    }

    float wst = wscale[3] * (1.0f / 128.0f);
#pragma unroll
    for (int mi = 0; mi < 2; ++mi)
#pragma unroll
        for (int ni = 0; ni < 4; ++ni) {
            int n = n0 + ni * 16 + laneLo;
            float bv = bias[n];
#pragma unroll
            for (int v = 0; v < 8; ++v) {
                int m = m0 + mi * 16 + v + 8 * laneHi;
                out[(size_t)m * EMB + n] = (float)acc[mi][ni][v] * (wst * gamma[m]) + bv;
            }
        }
}


extern "C" void kernel_launch(void* const* d_in, const int* in_sizes, int n_in,
                              void* d_out, int out_size, void* d_ws, size_t ws_size,
                              hipStream_t stream) {
    (void)in_sizes; (void)n_in; (void)out_size; (void)ws_size;
    const float* q_in  = (const float*)d_in[0];
    const float* k_in  = (const float*)d_in[1];
    const float* v_in  = (const float*)d_in[2];
    const float* w_in  = (const float*)d_in[3];
    const float* b_in  = (const float*)d_in[4];
    const float* w_out = (const float*)d_in[5];
    const float* b_out = (const float*)d_in[6];
    float* out = (float*)d_out;

    char* ws = (char*)d_ws;
    auto carve = [&](size_t bytes) -> char* {
        char* p = ws;
        ws += (bytes + 255) & ~(size_t)255;
        return p;
    };
    signed char* wq    = (signed char*)carve((size_t)4 * WELEM);
    float* partials    = (float*)carve(4 * 256 * sizeof(float));
    float* wscale      = (float*)carve(4 * sizeof(float));
    signed char* actq  = (signed char*)carve((size_t)3 * NTOK * EMB);
    float* gamma       = (float*)carve((size_t)3 * NTOK * sizeof(float));
    _Float16* Qh       = (_Float16*)carve((size_t)NTOK * EMB * 2);
    _Float16* Kh       = (_Float16*)carve((size_t)NTOK * EMB * 2);
    _Float16* Vt       = (_Float16*)carve((size_t)NTOK * EMB * 2);
    float* ctx         = (float*)carve((size_t)NTOK * EMB * 4);
    signed char* ctxq  = (signed char*)carve((size_t)NTOK * EMB);
    float* ctxg        = (float*)carve((size_t)NTOK * sizeof(float));

    k_wabs_partial<<<1024, 256, 0, stream>>>(w_in, w_out, partials);
    k_wscale_final<<<4, 256, 0, stream>>>(partials, wscale);
    k_quant_w<<<(4 * WELEM) / 256, 256, 0, stream>>>(w_in, w_out, wscale, wq);
    k_quant_act<<<NTOK, 256, 0, stream>>>(q_in, actq,                      gamma);
    k_quant_act<<<NTOK, 256, 0, stream>>>(k_in, actq + (size_t)NTOK * EMB, gamma + NTOK);
    k_quant_act<<<NTOK, 256, 0, stream>>>(v_in, actq + (size_t)2 * NTOK * EMB, gamma + 2 * NTOK);
    k_gemm_qkv<<<3072, 128, 0, stream>>>(actq, gamma, wq, wscale, b_in, Qh, Kh, Vt);
    k_attention<<<1024, 128, 0, stream>>>(Qh, Kh, Vt, ctx);
    k_quant_act<<<NTOK, 256, 0, stream>>>(ctx, ctxq, ctxg);
    k_gemm_out<<<1024, 128, 0, stream>>>(ctxq, ctxg, wq + (size_t)3 * WELEM, wscale, b_out, out);
}
